// LogicGraphNetwork_81003083202898
// MI455X (gfx1250) — matrix operand order repaired and checked by reading, not compiled, not run on hardware
//
#include <hip/hip_runtime.h>

// ---------------------------------------------------------------------------
// CDNA5 (gfx1250) wave32 + WMMA bf16 implementation of the message-passing GNN.
// All GEMMs computed transposed:  C[out16 x n16] = Wt_frag(A) * act_t(B) + bias
//   A (weights, 16x32 bf16): pre-swizzled fragments (prep_kernel)
//   B (activations, 32x16 bf16): lane = column (edge/node), 16 contiguous K/lane
//   C (f32 16x16): lane<16 -> rows 0..7, lane>=16 -> rows 8..15
// msg_kernel processes 2 edge-tiles per wave so each A fragment load feeds
// two WMMAs (halves the dominant weight-fragment fetch traffic).
// ---------------------------------------------------------------------------

typedef __attribute__((ext_vector_type(16))) __bf16 v16bf;
typedef __attribute__((ext_vector_type(8)))  float  v8f;

#define N_NODES 50000
#define N_EDGES 800000

// ---- small helpers --------------------------------------------------------

__device__ __forceinline__ v16bf ldv16(const __bf16* p) {
  return *(const v16bf*)p;                     // 32B, 32B-aligned -> 2x b128
}

__device__ __forceinline__ v16bf cvt16(const float* p) {
  v16bf r;
#pragma unroll
  for (int i = 0; i < 16; ++i) r[i] = (__bf16)p[i];
  return r;
}

__device__ __forceinline__ v8f wmma_bf16(v16bf A, v16bf B, v8f C) {
  return __builtin_amdgcn_wmma_f32_16x16x32_bf16(false, A, false, B, (short)0, C,
                                                 false, false);
}

__device__ __forceinline__ void init_bias(v8f acc[4], const float* b, int lane) {
  const int o = (lane & 16) ? 8 : 0;
#pragma unroll
  for (int mt = 0; mt < 4; ++mt)
#pragma unroll
    for (int r = 0; r < 8; ++r) acc[mt][r] = b[mt * 16 + o + r];
}

template <int NKT>
__device__ __forceinline__ void gemm_acc(v8f acc[4], const v16bf B[NKT],
                                         const __bf16* __restrict__ frags, int lane) {
#pragma unroll
  for (int mt = 0; mt < 4; ++mt)
#pragma unroll
    for (int kt = 0; kt < NKT; ++kt) {
      v16bf A = ldv16(frags + (mt * NKT + kt) * 512 + lane * 16);
      acc[mt] = wmma_bf16(A, B[kt], acc[mt]);
    }
}

union pack8 { __bf16 h[8]; uint4 u; };

// Write the 64-wide result row for this lane's column (edge/node) as bf16.
// row must already point at (buf + col*64).
__device__ __forceinline__ void store_rows_bf16(__bf16* row, const v8f acc[4],
                                                int lane, bool relu) {
  const int co = (lane & 16) ? 8 : 0;
#pragma unroll
  for (int mt = 0; mt < 4; ++mt) {
    pack8 pk;
#pragma unroll
    for (int r = 0; r < 8; ++r) {
      float v = acc[mt][r];
      if (relu) v = fmaxf(v, 0.f);
      pk.h[r] = (__bf16)v;
    }
    *(uint4*)(row + mt * 16 + co) = pk.u;      // b128 store
  }
}

// ---- weight fragment preparation ------------------------------------------
// W is [K][64] row-major f32.  Fragment (mt,kt) holds Wt rows mt*16..+15,
// K slice kt*32..+31, in the 16-bit WMMA A layout, lane-contiguous.
__global__ void prep_kernel(const float* __restrict__ W, __bf16* __restrict__ F, int K) {
  const int nkt   = K >> 5;
  const int total = nkt * 4 * 512;
  const int t = blockIdx.x * blockDim.x + threadIdx.x;
  if (t >= total) return;
  const int f    = t >> 9;
  const int mt   = f / nkt;
  const int kt   = f - mt * nkt;
  const int el   = t & 511;
  const int lane = el >> 4;
  const int j    = el & 15;
  const int m     = mt * 16 + (lane & 15);
  const int kbase = kt * 32 + ((lane & 16) ? 16 : 0);
  const int k     = kbase + j;                         // {0..15} / {16..31}
  F[t] = (__bf16)W[k * 64 + m];
}

// ---- encoders -------------------------------------------------------------

__global__ void enc_node_kernel(const float* __restrict__ nf, const __bf16* __restrict__ Wf,
                                const float* __restrict__ b, __bf16* __restrict__ xout) {
  const int wv = threadIdx.x >> 5, lane = threadIdx.x & 31;
  const int tile = blockIdx.x * 8 + wv;
  if (tile * 16 >= N_NODES) return;
  const int n  = tile * 16 + (lane & 15);
  const int hb = (lane & 16) ? 16 : 0;
  const float* row = nf + (size_t)n * 128;
  v16bf B[4];
#pragma unroll
  for (int kt = 0; kt < 4; ++kt) B[kt] = cvt16(row + kt * 32 + hb);
  v8f acc[4];
  init_bias(acc, b, lane);
  gemm_acc<4>(acc, B, Wf, lane);
  store_rows_bf16(xout + (size_t)n * 64, acc, lane, false);
}

__global__ void enc_edge_kernel(const float* __restrict__ ef, const __bf16* __restrict__ Wf,
                                const float* __restrict__ b, __bf16* __restrict__ eout) {
  const int wv = threadIdx.x >> 5, lane = threadIdx.x & 31;
  const int tile = blockIdx.x * 8 + wv;          // E divisible by 128: no guard
  const int n  = tile * 16 + (lane & 15);
  const int hb = (lane & 16) ? 16 : 0;
  const float* row = ef + (size_t)n * 64;
  v16bf B[2];
#pragma unroll
  for (int kt = 0; kt < 2; ++kt) B[kt] = cvt16(row + kt * 32 + hb);
  v8f acc[4];
  init_bias(acc, b, lane);
  gemm_acc<2>(acc, B, Wf, lane);
  store_rows_bf16(eout + (size_t)n * 64, acc, lane, false);
}

// ---- edge message MLP + scatter-add (2 tiles = 32 edges per wave) ---------

__global__ void msg_kernel(const __bf16* __restrict__ x, const __bf16* __restrict__ e,
                           const int* __restrict__ ei,
                           const __bf16* __restrict__ w1f, const float* __restrict__ b1,
                           const __bf16* __restrict__ w2f, const float* __restrict__ b2,
                           float* __restrict__ agg) {
  __shared__ __bf16 hsm[8 * 2 * 16 * 64];        // per wave: 2 x (16x64) bf16 h-tiles
  const int wv = threadIdx.x >> 5, lane = threadIdx.x & 31;
  const int base = (blockIdx.x * 8 + wv) * 32;   // 32 edges per wave
  const int hb = (lane & 16) ? 16 : 0;

  int dst[2];
  v16bf B[2][6];
#pragma unroll
  for (int t = 0; t < 2; ++t) {
    const int eL = base + t * 16 + (lane & 15);
    const int src = ei[eL];
    dst[t] = ei[N_EDGES + eL];
    const __bf16* xs = x + (size_t)src * 64;
    const __bf16* er = e + (size_t)eL * 64;
    const __bf16* xd = x + (size_t)dst[t] * 64;
    B[t][0] = ldv16(xs + 0 + hb);  B[t][1] = ldv16(xs + 32 + hb);
    B[t][2] = ldv16(er + 0 + hb);  B[t][3] = ldv16(er + 32 + hb);
    B[t][4] = ldv16(xd + 0 + hb);  B[t][5] = ldv16(xd + 32 + hb);
  }

  v8f acc[2][4];
  init_bias(acc[0], b1, lane);
  init_bias(acc[1], b1, lane);
#pragma unroll
  for (int mt = 0; mt < 4; ++mt)
#pragma unroll
    for (int kt = 0; kt < 6; ++kt) {
      v16bf A = ldv16(w1f + (mt * 6 + kt) * 512 + lane * 16);  // 1 load, 2 WMMAs
      acc[0][mt] = wmma_bf16(A, B[0][kt], acc[0][mt]);
      acc[1][mt] = wmma_bf16(A, B[1][kt], acc[1][mt]);
    }

  // relu -> bf16 -> LDS (C layout -> per-edge rows), then reload as B layout
#pragma unroll
  for (int t = 0; t < 2; ++t) {
    __bf16* row = hsm + (wv * 2 + t) * 1024 + (lane & 15) * 64;
    store_rows_bf16(row, acc[t], lane, true);
  }

  v16bf B2[2][2];
#pragma unroll
  for (int t = 0; t < 2; ++t) {
    const __bf16* row = hsm + (wv * 2 + t) * 1024 + (lane & 15) * 64;
    B2[t][0] = ldv16(row + 0 + hb);
    B2[t][1] = ldv16(row + 32 + hb);
  }

  v8f acc2[2][4];
  init_bias(acc2[0], b2, lane);
  init_bias(acc2[1], b2, lane);
#pragma unroll
  for (int mt = 0; mt < 4; ++mt)
#pragma unroll
    for (int kt = 0; kt < 2; ++kt) {
      v16bf A = ldv16(w2f + (mt * 2 + kt) * 512 + lane * 16);
      acc2[0][mt] = wmma_bf16(A, B2[0][kt], acc2[0][mt]);
      acc2[1][mt] = wmma_bf16(A, B2[1][kt], acc2[1][mt]);
    }

  // scatter-add m into agg[dst] (L2-resident, native f32 atomics)
  const int co = (lane & 16) ? 8 : 0;
#pragma unroll
  for (int t = 0; t < 2; ++t) {
    float* arow = agg + (size_t)dst[t] * 64;
#pragma unroll
    for (int mt = 0; mt < 4; ++mt)
#pragma unroll
      for (int r = 0; r < 8; ++r)
        unsafeAtomicAdd(arow + mt * 16 + co + r, acc2[t][mt][r]);
  }
}

// ---- node update MLP ------------------------------------------------------

__global__ void upd_kernel(const __bf16* __restrict__ x, const float* __restrict__ agg,
                           const __bf16* __restrict__ w1f, const float* __restrict__ b1,
                           const __bf16* __restrict__ w2f, const float* __restrict__ b2,
                           __bf16* __restrict__ xout, float* __restrict__ xf32) {
  __shared__ __bf16 hsm[8 * 16 * 64];
  const int wv = threadIdx.x >> 5, lane = threadIdx.x & 31;
  const int tile = blockIdx.x * 8 + wv;
  if (tile * 16 >= N_NODES) return;
  const int n  = tile * 16 + (lane & 15);
  const int hb = (lane & 16) ? 16 : 0;

  // upd_in^T: K 0..63 = x (bf16), 64..127 = agg (f32 -> bf16)
  v16bf B[4];
  const __bf16* xr = x + (size_t)n * 64;
  const float*  ar = agg + (size_t)n * 64;
  B[0] = ldv16(xr + 0 + hb);  B[1] = ldv16(xr + 32 + hb);
  B[2] = cvt16(ar + 0 + hb);  B[3] = cvt16(ar + 32 + hb);

  v8f acc[4];
  init_bias(acc, b1, lane);
  gemm_acc<4>(acc, B, w1f, lane);

  __bf16* row = hsm + wv * 1024 + (lane & 15) * 64;
  store_rows_bf16(row, acc, lane, true);

  v16bf B2[2];
  B2[0] = ldv16(row + 0 + hb);
  B2[1] = ldv16(row + 32 + hb);

  v8f acc2[4];
  init_bias(acc2, b2, lane);
  gemm_acc<2>(acc2, B2, w2f, lane);

  store_rows_bf16(xout + (size_t)n * 64, acc2, lane, false);

  if (xf32) {                                    // final layer: f32 x into d_out
    const int co = (lane & 16) ? 8 : 0;
    float* orow = xf32 + (size_t)n * 64;
#pragma unroll
    for (int mt = 0; mt < 4; ++mt)
      *(v8f*)(orow + mt * 16 + co) = acc2[mt];   // 32B contiguous -> 2x b128
  }
}

// ---- constraint head (tiny, plain VALU) -----------------------------------

__global__ void con_kernel(const float* __restrict__ x, const float* __restrict__ W1,
                           const float* __restrict__ b1, const float* __restrict__ W2,
                           const float* __restrict__ b2, float* __restrict__ out) {
  const int n = blockIdx.x * blockDim.x + threadIdx.x;
  if (n >= N_NODES) return;
  const float* xr = x + (size_t)n * 64;
  float xl[64];
#pragma unroll
  for (int k = 0; k < 64; ++k) xl[k] = xr[k];
  float s = b2[0];
  for (int j = 0; j < 64; ++j) {                 // W1 column access is uniform -> SMEM
    float h = b1[j];
#pragma unroll
    for (int k = 0; k < 64; ++k) h = fmaf(xl[k], W1[k * 64 + j], h);
    s = fmaf(fmaxf(h, 0.f), W2[j], s);
  }
  out[n] = 1.f / (1.f + expf(-s));
}

// ---------------------------------------------------------------------------

extern "C" void kernel_launch(void* const* d_in, const int* in_sizes, int n_in,
                              void* d_out, int out_size, void* d_ws, size_t ws_size,
                              hipStream_t stream) {
  const float* node_f = (const float*)d_in[0];
  const float* edge_f = (const float*)d_in[1];
  const int*   ei     = (const int*)d_in[2];
  const float* encNW  = (const float*)d_in[3];
  const float* encNb  = (const float*)d_in[4];
  const float* encEW  = (const float*)d_in[5];
  const float* encEb  = (const float*)d_in[6];
  const float* mW1    = (const float*)d_in[7];
  const float* mb1    = (const float*)d_in[8];
  const float* mW2    = (const float*)d_in[9];
  const float* mb2    = (const float*)d_in[10];
  const float* uW1    = (const float*)d_in[11];
  const float* ub1    = (const float*)d_in[12];
  const float* uW2    = (const float*)d_in[13];
  const float* ub2    = (const float*)d_in[14];
  const float* cW1    = (const float*)d_in[15];
  const float* cb1    = (const float*)d_in[16];
  const float* cW2    = (const float*)d_in[17];
  const float* cb2    = (const float*)d_in[18];

  char*  ws  = (char*)d_ws;
  size_t off = 0;
  auto alloc = [&](size_t bytes) -> void* {
    void* p = ws + off;
    off = (off + bytes + 255) & ~(size_t)255;
    return p;
  };
  __bf16* encNf = (__bf16*)alloc(16 * 1024);           // 4kt x 4mt frags
  __bf16* encEf = (__bf16*)alloc(8 * 1024);            // 2kt x 4mt
  __bf16* mW1f  = (__bf16*)alloc(3 * 24 * 1024);       // per layer 6kt x 4mt
  __bf16* mW2f  = (__bf16*)alloc(3 * 8 * 1024);
  __bf16* uW1f  = (__bf16*)alloc(3 * 16 * 1024);
  __bf16* uW2f  = (__bf16*)alloc(3 * 8 * 1024);
  __bf16* x0    = (__bf16*)alloc((size_t)N_NODES * 64 * 2);
  __bf16* x1    = (__bf16*)alloc((size_t)N_NODES * 64 * 2);
  __bf16* ebuf  = (__bf16*)alloc((size_t)N_EDGES * 64 * 2);
  float*  agg   = (float*)alloc((size_t)N_NODES * 64 * 4);

  auto prep = [&](const float* W, __bf16* F, int K) {
    int total = (K / 32) * 4 * 512;
    prep_kernel<<<(total + 255) / 256, 256, 0, stream>>>(W, F, K);
  };
  prep(encNW, encNf, 128);
  prep(encEW, encEf, 64);
  for (int l = 0; l < 3; ++l) {
    prep(mW1 + (size_t)l * 192 * 64, mW1f + (size_t)l * 12288, 192);
    prep(mW2 + (size_t)l * 64 * 64,  mW2f + (size_t)l * 4096,  64);
    prep(uW1 + (size_t)l * 128 * 64, uW1f + (size_t)l * 8192,  128);
    prep(uW2 + (size_t)l * 64 * 64,  uW2f + (size_t)l * 4096,  64);
  }

  const int nodeBlocks = (N_NODES / 16 + 7) / 8;       // 3125 wave-tiles
  const int encEdgeBlocks = N_EDGES / 128;             // exact: 6250
  const int msgBlocks     = N_EDGES / 256;             // exact: 3125 (32 edges/wave)

  enc_node_kernel<<<nodeBlocks, 256, 0, stream>>>(node_f, encNf, encNb, x0);
  enc_edge_kernel<<<encEdgeBlocks, 256, 0, stream>>>(edge_f, encEf, encEb, ebuf);

  float* outx = (float*)d_out;                         // [N,64] f32
  float* outc = outx + (size_t)N_NODES * 64;           // [N,1]  f32

  __bf16* xc = x0; __bf16* xn = x1;
  for (int l = 0; l < 3; ++l) {
    hipMemsetAsync(agg, 0, (size_t)N_NODES * 64 * 4, stream);
    msg_kernel<<<msgBlocks, 256, 0, stream>>>(
        xc, ebuf, ei,
        mW1f + (size_t)l * 12288, mb1 + (size_t)l * 64,
        mW2f + (size_t)l * 4096,  mb2 + (size_t)l * 64, agg);
    upd_kernel<<<nodeBlocks, 256, 0, stream>>>(
        xc, agg,
        uW1f + (size_t)l * 8192, ub1 + (size_t)l * 64,
        uW2f + (size_t)l * 4096, ub2 + (size_t)l * 64,
        xn, (l == 2) ? outx : (float*)nullptr);
    __bf16* t = xc; xc = xn; xn = t;
  }

  con_kernel<<<(N_NODES + 255) / 256, 256, 0, stream>>>(outx, cW1, cb1, cW2, cb2, outc);

  (void)in_sizes; (void)n_in; (void)out_size; (void)ws_size;
}
